// GAEENcoder_11029476016716
// MI455X (gfx1250) — compile-verified, matrix operand order checked by reading, not run on hardware
//
#include <hip/hip_runtime.h>
#include <hip/hip_bf16.h>
#include <stdint.h>

typedef __attribute__((ext_vector_type(16))) __bf16 v16bf;
typedef __attribute__((ext_vector_type(8)))  float  v8f;

// ---------- helpers ----------
__device__ __forceinline__ __bf16 f2bf(float f) {
    union { float f; unsigned u; } v; v.f = f;
    unsigned r = v.u + 0x7FFFu + ((v.u >> 16) & 1u);   // round-to-nearest-even
    unsigned short h = (unsigned short)(r >> 16);
    return __builtin_bit_cast(__bf16, h);
}

// ---------- degree / dinv (graph norm computed once, reused by both layers) ----------
__global__ void k_deg_init(unsigned* __restrict__ deg, int n) {
    int i = blockIdx.x * blockDim.x + threadIdx.x;
    if (i < n) deg[i] = 1u;                            // self-loop
}

__global__ void k_deg_edges(const long long* __restrict__ dst, unsigned* __restrict__ deg,
                            long long E) {
    long long i = (long long)blockIdx.x * blockDim.x + threadIdx.x;
    if (i < E) atomicAdd(&deg[(int)dst[i]], 1u);
}

__global__ void k_dinv(const unsigned* __restrict__ deg, float* __restrict__ dinv, int n) {
    int i = blockIdx.x * blockDim.x + threadIdx.x;
    if (i < n) dinv[i] = rsqrtf((float)deg[i]);
}

// ---------- fused WMMA GEMM ----------
// H[M,N] = act(A[M,K]) * W[K,N]   (act = BN+ReLU if BN, else identity)
// OutInit[M,N] = H * dinv[row]^2 + bias[col]   (self-loop + bias epilogue)
// One wave owns a 16-row strip x full N; W staged once per block as bf16 in LDS.
template <int K, int N, bool BN>
__global__ void k_gemm_fused(const float* __restrict__ A, const float* __restrict__ W,
                             const float* __restrict__ bnsc, const float* __restrict__ bnsh,
                             const float* __restrict__ dinv, const float* __restrict__ bias,
                             float* __restrict__ H, float* __restrict__ OutInit, int M) {
    constexpr int PITCH = K + 2;        // bf16 pitch: keeps column stride off one bank
    constexpr int NT = N / 16;
    constexpr int KS = K / 32;
    __shared__ __bf16 lw[N * PITCH];    // W^T [N][K+2] in bf16
    __shared__ float  lsc[K], lsh[K], lbias[N];

    // cooperative staging (coalesced global reads of row-major W)
    for (int idx = threadIdx.x; idx < K * N; idx += blockDim.x) {
        int k = idx / N, n = idx % N;
        lw[n * PITCH + k] = f2bf(W[idx]);
    }
    if constexpr (BN) {
        for (int c = threadIdx.x; c < K; c += blockDim.x) { lsc[c] = bnsc[c]; lsh[c] = bnsh[c]; }
    }
    for (int c = threadIdx.x; c < N; c += blockDim.x) lbias[c] = bias[c];
    __syncthreads();

    const int lane  = threadIdx.x & 31;
    const int wave  = threadIdx.x >> 5;
    const int mtile = blockIdx.x * (blockDim.x >> 5) + wave;
    if (mtile >= M / 16) return;        // uniform per wave -> EXEC all-1s for WMMA
    const int tm   = mtile * 16;
    const int r    = lane & 15;
    const int half = (lane < 16) ? 0 : 8;   // K sub-block / M-offset per lane half

    // A fragments, loaded once, reused across all N tiles (ISA 16-bit A layout)
    v16bf af[KS];
    const float* ap = A + (size_t)(tm + r) * K;
#pragma unroll
    for (int kk = 0; kk < KS; ++kk) {
        const int k0 = kk * 32;
#pragma unroll
        for (int e = 0; e < 8; ++e) {
            const int c0 = k0 + half + e;
            const int c1 = k0 + 16 + half + e;
            float v0 = ap[c0], v1 = ap[c1];
            if constexpr (BN) {
                v0 = v0 * lsc[c0] + lsh[c0]; v0 = v0 > 0.f ? v0 : 0.f;
                v1 = v1 * lsc[c1] + lsh[c1]; v1 = v1 > 0.f ? v1 : 0.f;
            }
            af[kk][e]     = f2bf(v0);
            af[kk][e + 8] = f2bf(v1);
        }
    }
    // dinv^2 for the 8 output rows this lane-half stores
    float d2[8];
#pragma unroll
    for (int v = 0; v < 8; ++v) { float d = dinv[tm + v + half]; d2[v] = d * d; }

#pragma unroll
    for (int j = 0; j < NT; ++j) {
        v8f acc = {};
        const __bf16* wp = &lw[(j * 16 + r) * PITCH];
#pragma unroll
        for (int kk = 0; kk < KS; ++kk) {
            v16bf b;
            const __bf16* w0 = wp + kk * 32 + half;   // 8 contiguous bf16 (16B ds load)
#pragma unroll
            for (int e = 0; e < 8; ++e) { b[e] = w0[e]; b[e + 8] = w0[16 + e]; }
            acc = __builtin_amdgcn_wmma_f32_16x16x32_bf16(
                false, af[kk], false, b, (short)0, acc, false, false);
        }
        // epilogue: C/D layout -> row = tm + v + half, col = j*16 + r
        const int col = j * 16 + r;
        const float bb = lbias[col];
        float* hp = H + (size_t)tm * N + col;
        float* op = OutInit + (size_t)tm * N + col;
#pragma unroll
        for (int v = 0; v < 8; ++v) {
            float hv = acc[v];
            hp[(size_t)(v + half) * N] = hv;
            op[(size_t)(v + half) * N] = hv * d2[v] + bb;
        }
    }
}

// ---------- edge scatter-add: one wave per edge, VEC floats per lane ----------
template <int VEC>
__global__ void k_agg_edges(const long long* __restrict__ src, const long long* __restrict__ dst,
                            const float* __restrict__ dinv, const float* __restrict__ h,
                            float* __restrict__ out, long long E, int Cc) {
    const int lane = threadIdx.x & 31;
    long long wid = (long long)blockIdx.x * (blockDim.x >> 5) + (threadIdx.x >> 5);
    long long nw  = (long long)gridDim.x * (blockDim.x >> 5);
    for (long long e = wid; e < E; e += nw) {
        int s = (int)src[e];
        int d = (int)dst[e];
        float nrm = dinv[s] * dinv[d];
        const float* hp = h + (size_t)s * Cc + lane * VEC;   // contiguous row gather
        float* op = out + (size_t)d * Cc + lane * VEC;
#pragma unroll
        for (int v = 0; v < VEC; ++v)
            atomicAdd(op + v, hp[v] * nrm);                  // L2 atomics (fits 192MB L2)
    }
}

// ---------- batch-norm stats (coalesced: 256 thr = 2 rows x 128 ch) ----------
__global__ void k_zero(float* __restrict__ p, int n) {
    int i = blockIdx.x * blockDim.x + threadIdx.x;
    if (i < n) p[i] = 0.f;
}

__global__ void k_bn_partial(const float* __restrict__ x, float* __restrict__ sums, int Mrows) {
    const int C = 128;
    const int c  = threadIdx.x & (C - 1);
    const int rh = threadIdx.x >> 7;
    float s = 0.f, s2 = 0.f;
    for (int row = blockIdx.x * 2 + rh; row < Mrows; row += gridDim.x * 2) {
        float v = x[(size_t)row * C + c];
        s += v; s2 += v * v;
    }
    __shared__ float red[256];
    red[threadIdx.x] = s;
    __syncthreads();
    if (threadIdx.x < C) atomicAdd(&sums[c], red[c] + red[c + C]);
    __syncthreads();
    red[threadIdx.x] = s2;
    __syncthreads();
    if (threadIdx.x < C) atomicAdd(&sums[C + c], red[c] + red[c + C]);
}

__global__ void k_bn_final(const float* __restrict__ sums, const float* __restrict__ gamma,
                           const float* __restrict__ beta, float* __restrict__ bnsc,
                           float* __restrict__ bnsh, int Mrows) {
    int c = threadIdx.x;
    if (c < 128) {
        float m   = sums[c] / (float)Mrows;
        float var = sums[128 + c] / (float)Mrows - m * m;
        float sc  = rsqrtf(var + 1e-5f) * gamma[c];
        bnsc[c] = sc;
        bnsh[c] = beta[c] - m * sc;
    }
}

// ---------- host launch ----------
extern "C" void kernel_launch(void* const* d_in, const int* in_sizes, int n_in,
                              void* d_out, int out_size, void* d_ws, size_t ws_size,
                              hipStream_t stream) {
    (void)n_in; (void)out_size; (void)ws_size;
    const float*     x     = (const float*)d_in[0];
    const long long* ei    = (const long long*)d_in[1];   // int64 [2, E]
    const float*     W1    = (const float*)d_in[2];
    const float*     b1    = (const float*)d_in[3];
    const float*     gamma = (const float*)d_in[4];
    const float*     beta  = (const float*)d_in[5];
    const float*     W2    = (const float*)d_in[6];
    const float*     b2    = (const float*)d_in[7];
    float* out = (float*)d_out;

    const int IN = 64, HID = 128, LAT = 64;
    const int M = in_sizes[0] / IN;                        // 50000
    const long long E = (long long)(in_sizes[1] / 2);      // 1.6M
    const long long* src = ei;
    const long long* dst = ei + E;

    // workspace layout (256B aligned slices)
    char* ws = (char*)d_ws;
    size_t off = 0;
    auto alloc = [&](size_t bytes) { char* p = ws + off; off = (off + bytes + 255) & ~(size_t)255; return p; };
    unsigned* deg  = (unsigned*)alloc((size_t)M * 4);
    float*    dinv = (float*)   alloc((size_t)M * 4);
    float*    h1   = (float*)   alloc((size_t)M * HID * 4);   // layer-1 pre-agg features
    float*    out1 = (float*)   alloc((size_t)M * HID * 4);   // layer-1 aggregation
    float*    h2   = (float*)   alloc((size_t)M * LAT * 4);   // layer-2 pre-agg features
    float*    sums = (float*)   alloc(2 * HID * 4);           // [sum | sumsq]
    float*    bnsc = (float*)   alloc(HID * 4);
    float*    bnsh = (float*)   alloc(HID * 4);

    const int T = 256;
    const int gemmBlocks = (M / 16 + 7) / 8;               // 8 waves/block, 1 M-strip/wave

    // --- graph normalization ---
    k_deg_init<<<(M + T - 1) / T, T, 0, stream>>>(deg, M);
    k_deg_edges<<<(int)((E + T - 1) / T), T, 0, stream>>>(dst, deg, E);
    k_dinv<<<(M + T - 1) / T, T, 0, stream>>>(deg, dinv, M);

    // --- layer 1: fused GEMM(+self-loop+bias epilogue) -> edge scatter ---
    k_gemm_fused<64, 128, false><<<gemmBlocks, T, 0, stream>>>(
        x, W1, nullptr, nullptr, dinv, b1, h1, out1, M);
    k_agg_edges<4><<<(int)((E + 7) / 8), T, 0, stream>>>(src, dst, dinv, h1, out1, E, HID);

    // --- batch norm stats (BN+ReLU itself fused into GEMM2 A-load) ---
    k_zero<<<1, 2 * HID, 0, stream>>>(sums, 2 * HID);
    k_bn_partial<<<256, T, 0, stream>>>(out1, sums, M);
    k_bn_final<<<1, HID, 0, stream>>>(sums, gamma, beta, bnsc, bnsh, M);

    // --- layer 2: fused BN+ReLU -> GEMM(+self-loop+bias into d_out) -> edge scatter ---
    k_gemm_fused<128, 64, true><<<gemmBlocks, T, 0, stream>>>(
        out1, W2, bnsc, bnsh, dinv, b2, h2, out, M);
    k_agg_edges<2><<<(int)((E + 7) / 8), T, 0, stream>>>(src, dst, dinv, h2, out, E, LAT);
}
